// CraftingGraph_36180804501653
// MI455X (gfx1250) — matrix operand order repaired and checked by reading, not compiled, not run on hardware
//
#include <hip/hip_runtime.h>
#include <cstddef>

#define NCOL 128   // HID == EMB == 128

typedef __attribute__((ext_vector_type(16))) __bf16 v16bf;
typedef __attribute__((ext_vector_type(8)))  float  v8f;

struct U16x16 { unsigned short u[16]; };

__device__ __forceinline__ unsigned short f2bf(float f) {
    unsigned int u = __float_as_uint(f);
    u += 0x7FFFu + ((u >> 16) & 1u);          // round-to-nearest-even
    return (unsigned short)(u >> 16);
}

__device__ __forceinline__ void atomicMaxF(float* addr, float val) {
    if (val >= 0.0f) atomicMax((int*)addr, __float_as_int(val));
    else             atomicMin((unsigned int*)addr, __float_as_uint(val));
}

// ---------------------------------------------------------------------------
// GEMM: H = bf16(X) @ bf16(W), plus S = H@a_src, D = H@a_dst.
// X: (8192, K) f32 row-major, W: (K, 128) f32 row-major.
// One block = 128 rows; 8 waves, each wave owns a 16-row strip and walks the
// 8 column tiles with v_wmma_f32_16x16x32_bf16.
// ---------------------------------------------------------------------------
template <int K>
__global__ __launch_bounds__(256)
void gat_gemm(const float* __restrict__ X, const float* __restrict__ W,
              const float* __restrict__ asrc, const float* __restrict__ adst,
              float* __restrict__ H, float* __restrict__ S, float* __restrict__ D)
{
    __shared__ unsigned short sX[128 * K];   // [localRow][k]  bf16
    __shared__ unsigned short sW[128 * K];   // [n][k] (W transposed) bf16

    const int tid      = threadIdx.x;
    const int blockRow = blockIdx.x * 128;

    // Stage W transposed: sW[n*K + k] = bf16(W[k*128 + n])
    for (int idx = tid; idx < K * 128; idx += 256) {
        const int k = idx >> 7;
        const int n = idx & 127;
        sW[n * K + k] = f2bf(W[idx]);
    }
    // Stage the 128-row X tile (contiguous), with speculative prefetch ahead.
    const float* Xblk = X + (size_t)blockRow * K;
    for (int idx = tid; idx < 128 * K; idx += 256) {
        __builtin_prefetch(Xblk + idx + 4096, 0, 1);   // global_prefetch_b8
        sX[idx] = f2bf(Xblk[idx]);
    }
    __syncthreads();

    const int wave = tid >> 5;
    const int lane = tid & 31;
    const int g    = lane >> 4;     // half-wave group (ISA A/B layout split)
    const int lm   = lane & 15;
    const int rowBase = wave * 16;  // local row tile

    constexpr int KS = K / 32;      // number of 16x16x32 steps

    // Preload A fragments for all K-steps (contiguous k order: element s holds k = 16*g + s, as for B).
    v16bf afrag[KS];
    #pragma unroll
    for (int ks = 0; ks < KS; ++ks) {
        U16x16 ta;
        const unsigned short* arow = &sX[(rowBase + lm) * K + ks * 32 + 16 * g];
        #pragma unroll
        for (int v = 0; v < 8; ++v) {
            const int kp = 2 * v;
            ta.u[2 * v]     = arow[kp];
            ta.u[2 * v + 1] = arow[kp + 1];
        }
        afrag[ks] = __builtin_bit_cast(v16bf, ta);
    }

    float sPart[8], dPart[8];
    #pragma unroll
    for (int r = 0; r < 8; ++r) { sPart[r] = 0.f; dPart[r] = 0.f; }

    for (int tn = 0; tn < 8; ++tn) {
        v8f c = {};
        #pragma unroll
        for (int ks = 0; ks < KS; ++ks) {
            // B 32x16 bf16: lane = column, K-pairs contiguous in sW (transposed).
            U16x16 tb;
            const unsigned short* bcol = &sW[(tn * 16 + lm) * K + ks * 32 + 16 * g];
            #pragma unroll
            for (int v = 0; v < 8; ++v) {
                tb.u[2 * v]     = bcol[2 * v];
                tb.u[2 * v + 1] = bcol[2 * v + 1];
            }
            const v16bf bfrag = __builtin_bit_cast(v16bf, tb);
            c = __builtin_amdgcn_wmma_f32_16x16x32_bf16(
                    false, afrag[ks], false, bfrag, (short)0, c, false, false);
        }
        // C layout: lane lm = column, VGPR r = row r + 8*g.
        const int   col  = tn * 16 + lm;
        const float av   = asrc[col];
        const float dv   = adst[col];
        float* Hrow = H + (size_t)(blockRow + rowBase + 8 * g) * NCOL + col;
        #pragma unroll
        for (int r = 0; r < 8; ++r) {
            Hrow[(size_t)r * NCOL] = c[r];
            sPart[r] += c[r] * av;
            dPart[r] += c[r] * dv;
        }
    }

    // Reduce the 16-lane partial dot products (wave32 shuffles, width=16).
    #pragma unroll
    for (int r = 0; r < 8; ++r) {
        float sv = sPart[r], dv = dPart[r];
        #pragma unroll
        for (int off = 8; off; off >>= 1) {
            sv += __shfl_xor(sv, off, 16);
            dv += __shfl_xor(dv, off, 16);
        }
        if (lm == 0) {
            const int row = blockRow + rowBase + r + 8 * g;
            S[row] = sv;
            D[row] = dv;
        }
    }
}

// ---------------------------------------------------------------------------
// Per-layer scratch init: ACC=0, rowmax=-inf, rowsum=0.
// ---------------------------------------------------------------------------
__global__ void init_layer(float* __restrict__ acc, float* __restrict__ rmax,
                           float* __restrict__ rsum, int n)
{
    const int i = blockIdx.x * blockDim.x + threadIdx.x;
    if (i < n * NCOL) acc[i] = 0.0f;
    if (i < n) { rmax[i] = -3.0e38f; rsum[i] = 0.0f; }
}

__device__ __forceinline__ float edge_score(const int* __restrict__ ei,
                                            const float* __restrict__ S,
                                            const float* __restrict__ D,
                                            int e, int E, int* dOut)
{
    const int s = ei[e];
    const int d = ei[E + e];
    *dOut = d;
    const float v = S[s] + D[d];
    return v > 0.0f ? v : 0.2f * v;   // leaky_relu(0.2)
}

__global__ void edge_max_kernel(const int* __restrict__ ei, const float* __restrict__ S,
                                const float* __restrict__ D, float* __restrict__ rmax, int E)
{
    const int e = blockIdx.x * blockDim.x + threadIdx.x;
    if (e >= E) return;
    int d; const float v = edge_score(ei, S, D, e, E, &d);
    atomicMaxF(&rmax[d], v);
}

__global__ void edge_sum_kernel(const int* __restrict__ ei, const float* __restrict__ S,
                                const float* __restrict__ D, const float* __restrict__ rmax,
                                float* __restrict__ rsum, int E)
{
    const int e = blockIdx.x * blockDim.x + threadIdx.x;
    if (e >= E) return;
    int d; const float v = edge_score(ei, S, D, e, E, &d);
    atomicAdd(&rsum[d], __expf(v - rmax[d]));
}

// One wave per edge: 32 lanes x 4 columns scatter-accumulate w * h[src].
__global__ void edge_scatter_kernel(const int* __restrict__ ei, const float* __restrict__ S,
                                    const float* __restrict__ D, const float* __restrict__ rmax,
                                    const float* __restrict__ rsum, const float* __restrict__ H,
                                    float* __restrict__ acc, int E)
{
    const int t    = blockIdx.x * blockDim.x + threadIdx.x;
    const int e    = t >> 5;
    const int lane = t & 31;
    if (e >= E) return;
    const int s = ei[e];
    const int d = ei[E + e];
    float v = S[s] + D[d];
    v = v > 0.0f ? v : 0.2f * v;
    const float w = __expf(v - rmax[d]) / rsum[d];
    if (w > 1e-6f) {                              // softmax threshold
        const float* hs = H   + (size_t)s * NCOL;
        float*       ad = acc + (size_t)d * NCOL;
        #pragma unroll
        for (int j = 0; j < 4; ++j) {
            const int col = lane + 32 * j;
            atomicAdd(&ad[col], w * hs[col]);
        }
    }
}

// ---------------------------------------------------------------------------
// Epilogue: y = elu(acc) + h ; out = layernorm(y) — one wave per row.
// ---------------------------------------------------------------------------
__global__ void post_kernel(const float* __restrict__ acc, const float* __restrict__ H,
                            const float* __restrict__ gamma, const float* __restrict__ beta,
                            float* __restrict__ out, int n)
{
    const int t    = blockIdx.x * blockDim.x + threadIdx.x;
    const int row  = t >> 5;
    const int lane = t & 31;
    if (row >= n) return;
    const float* a = acc + (size_t)row * NCOL;
    const float* h = H   + (size_t)row * NCOL;

    float y[4];
    float sum = 0.0f;
    #pragma unroll
    for (int j = 0; j < 4; ++j) {
        const int col = lane + 32 * j;
        const float x = a[col];
        const float el = x > 0.0f ? x : (__expf(x) - 1.0f);   // elu
        y[j] = el + h[col];
        sum += y[j];
    }
    #pragma unroll
    for (int off = 16; off; off >>= 1) sum += __shfl_xor(sum, off, 32);
    const float mu = sum * (1.0f / 128.0f);

    float var = 0.0f;
    #pragma unroll
    for (int j = 0; j < 4; ++j) { const float dd = y[j] - mu; var += dd * dd; }
    #pragma unroll
    for (int off = 16; off; off >>= 1) var += __shfl_xor(var, off, 32);
    var *= (1.0f / 128.0f);
    const float inv = rsqrtf(var + 1e-5f);

    float* o = out + (size_t)row * NCOL;
    #pragma unroll
    for (int j = 0; j < 4; ++j) {
        const int col = lane + 32 * j;
        o[col] = gamma[col] * (y[j] - mu) * inv + beta[col];
    }
}

// ---------------------------------------------------------------------------
extern "C" void kernel_launch(void* const* d_in, const int* in_sizes, int n_in,
                              void* d_out, int out_size, void* d_ws, size_t ws_size,
                              hipStream_t stream)
{
    const float* x0    = (const float*)d_in[0];
    const int*   ei    = (const int*)  d_in[1];   // (2,E) flat int32
    const float* W1    = (const float*)d_in[2];
    const float* asrc1 = (const float*)d_in[3];
    const float* adst1 = (const float*)d_in[4];
    const float* g1    = (const float*)d_in[5];
    const float* b1    = (const float*)d_in[6];
    const float* W2    = (const float*)d_in[7];
    const float* asrc2 = (const float*)d_in[8];
    const float* adst2 = (const float*)d_in[9];
    const float* g2    = (const float*)d_in[10];
    const float* b2    = (const float*)d_in[11];

    const int N = in_sizes[0] / 64;   // 8192
    const int E = in_sizes[1] / 2;    // 131072

    float* ws   = (float*)d_ws;
    float* S    = ws;                        // N
    float* D    = S    + N;                  // N
    float* rmax = D    + N;                  // N
    float* rsum = rmax + N;                  // N
    float* H    = rsum + N;                  // N*128
    float* ACC  = H    + (size_t)N * NCOL;   // N*128
    float* X1   = ACC  + (size_t)N * NCOL;   // N*128

    const dim3 blk(256);
    const int gridE   = (E + 255) / 256;
    const int gridE32 = (E * 32 + 255) / 256;
    const int gridN32 = (N * 32 + 255) / 256;
    const int gridNC  = (N * NCOL + 255) / 256;

    // ---- Layer 1 (K = 64) ----
    gat_gemm<64><<<N / 128, blk, 0, stream>>>(x0, W1, asrc1, adst1, H, S, D);
    init_layer<<<gridNC, blk, 0, stream>>>(ACC, rmax, rsum, N);
    edge_max_kernel<<<gridE, blk, 0, stream>>>(ei, S, D, rmax, E);
    edge_sum_kernel<<<gridE, blk, 0, stream>>>(ei, S, D, rmax, rsum, E);
    edge_scatter_kernel<<<gridE32, blk, 0, stream>>>(ei, S, D, rmax, rsum, H, ACC, E);
    post_kernel<<<gridN32, blk, 0, stream>>>(ACC, H, g1, b1, X1, N);

    // ---- Layer 2 (K = 128) ----
    gat_gemm<128><<<N / 128, blk, 0, stream>>>(X1, W2, asrc2, adst2, H, S, D);
    init_layer<<<gridNC, blk, 0, stream>>>(ACC, rmax, rsum, N);
    edge_max_kernel<<<gridE, blk, 0, stream>>>(ei, S, D, rmax, E);
    edge_sum_kernel<<<gridE, blk, 0, stream>>>(ei, S, D, rmax, rsum, E);
    edge_scatter_kernel<<<gridE32, blk, 0, stream>>>(ei, S, D, rmax, rsum, H, ACC, E);
    post_kernel<<<gridN32, blk, 0, stream>>>(ACC, H, g2, b2, (float*)d_out, N);
}
